// SparseMoEBlock_2267742732891
// MI455X (gfx1250) — matrix operand order repaired and checked by reading, not compiled, not run on hardware
//
#include <hip/hip_runtime.h>

#define D_MODEL 1024
#define HIDDEN  4096
#define N_EXP   8
#define T_TOK   2048

typedef __attribute__((ext_vector_type(16))) __bf16 v16bf;
typedef __attribute__((ext_vector_type(8)))  float  v8f;

// ---------------------------------------------------------------------------
// Kernel 0: zero the output accumulator and per-expert counters
// ---------------------------------------------------------------------------
__global__ void moe_zero(float* __restrict__ out, int n, int* __restrict__ cnt) {
    int i = blockIdx.x * blockDim.x + threadIdx.x;
    int stride = gridDim.x * blockDim.x;
    for (; i < n; i += stride) out[i] = 0.0f;
    if (blockIdx.x == 0 && threadIdx.x < N_EXP) cnt[threadIdx.x] = 0;
}

// ---------------------------------------------------------------------------
// Kernel 1: router. One block per token; wave e computes logit[t][e].
// ---------------------------------------------------------------------------
__global__ __launch_bounds__(256) void moe_router(
    const float* __restrict__ x, const float* __restrict__ rw,
    float* __restrict__ probs, float* __restrict__ wlist,
    int* __restrict__ list, int* __restrict__ cnt)
{
    const int t = blockIdx.x;
    const int wave = threadIdx.x >> 5;
    const int lane = threadIdx.x & 31;
    __shared__ float lg[N_EXP];

    const float* xr = x + (size_t)t * D_MODEL;
    const float* wr = rw + (size_t)wave * D_MODEL;
    float p = 0.0f;
    for (int j = lane; j < D_MODEL; j += 32) p += xr[j] * wr[j];
    for (int off = 16; off; off >>= 1) p += __shfl_down(p, off, 32);
    if (lane == 0) lg[wave] = p;
    __syncthreads();

    if (threadIdx.x == 0) {
        float mx = lg[0];
        for (int e = 1; e < N_EXP; ++e) mx = fmaxf(mx, lg[e]);
        float ex[N_EXP], Z = 0.0f;
        for (int e = 0; e < N_EXP; ++e) { ex[e] = __expf(lg[e] - mx); Z += ex[e]; }
        float invZ = 1.0f / Z;
        for (int e = 0; e < N_EXP; ++e) probs[t * N_EXP + e] = ex[e] * invZ;

        // top-2 with jax.top_k tie-breaking (earlier index wins on ties)
        int e0 = 0;
        for (int e = 1; e < N_EXP; ++e) if (lg[e] > lg[e0]) e0 = e;
        int e1 = -1;
        for (int e = 0; e < N_EXP; ++e) {
            if (e == e0) continue;
            if (e1 < 0 || lg[e] > lg[e1]) e1 = e;
        }
        float v0 = lg[e0], v1 = lg[e1];
        float m2 = fmaxf(v0, v1);
        float a = __expf(v0 - m2), b = __expf(v1 - m2);
        float inv = 1.0f / (a + b);
        float w0 = a * inv, w1c = b * inv;

        int p0 = atomicAdd(&cnt[e0], 1);
        list[e0 * T_TOK + p0] = t; wlist[e0 * T_TOK + p0] = w0;
        int p1 = atomicAdd(&cnt[e1], 1);
        list[e1 * T_TOK + p1] = t; wlist[e1 * T_TOK + p1] = w1c;
    }
}

// ---------------------------------------------------------------------------
// Kernel 2: load-balance aux loss (deterministic tree reduction)
// ---------------------------------------------------------------------------
__global__ __launch_bounds__(256) void moe_lbal(
    const float* __restrict__ probs, const int* __restrict__ cnt,
    float* __restrict__ out_scalar)
{
    __shared__ float red[256][N_EXP];
    float loc[N_EXP];
    for (int e = 0; e < N_EXP; ++e) loc[e] = 0.0f;
    for (int t = threadIdx.x; t < T_TOK; t += 256)
        for (int e = 0; e < N_EXP; ++e) loc[e] += probs[t * N_EXP + e];
    for (int e = 0; e < N_EXP; ++e) red[threadIdx.x][e] = loc[e];
    __syncthreads();
    for (int s = 128; s; s >>= 1) {
        if (threadIdx.x < (unsigned)s)
            for (int e = 0; e < N_EXP; ++e) red[threadIdx.x][e] += red[threadIdx.x + s][e];
        __syncthreads();
    }
    if (threadIdx.x == 0) {
        float l = 0.0f;
        for (int e = 0; e < N_EXP; ++e) {
            float f = (float)cnt[e] * (1.0f / T_TOK);
            float P = red[0][e] * (1.0f / T_TOK);
            l += f * P;
        }
        *out_scalar = (float)N_EXP * l;
    }
}

// ---------------------------------------------------------------------------
// Kernel 3: gathered expert GEMM, bf16x3 split-precision WMMA.
// Block tile: 64 gathered tokens x 64 H columns. 8 waves; wave w owns
// N-subtile (w&3) and M-subtiles (w>>2)*16 and (w>>2)*16+32  -> two 16x16
// f32 accumulators, 6 WMMAs per K-chunk (B fragments shared across M tiles).
// ---------------------------------------------------------------------------
__global__ __launch_bounds__(256, 2) void moe_gemm(
    const float* __restrict__ x, const float* __restrict__ w1,
    const float* __restrict__ b1,
    const int* __restrict__ list, const float* __restrict__ wlist,
    const int* __restrict__ cnt, float* __restrict__ out)
{
    const int e = blockIdx.z;
    const int count = cnt[e];
    const int m0 = blockIdx.y * 64;
    if (m0 >= count) return;                 // uniform early-exit
    const int n0 = blockIdx.x * 64;

    __shared__ __bf16 Ah[64][40], Al[64][40];   // 64 token-rows x 32 k (hi/lo)
    __shared__ __bf16 Bh[64][40], Bl[64][40];   // 64 h-rows    x 32 k (hi/lo)
    __shared__ float Wt[64];
    __shared__ int   Tok[64];

    const int tid = threadIdx.x;
    if (tid < 64) {
        int g = m0 + tid;
        if (g < count) { Tok[tid] = list[e * T_TOK + g]; Wt[tid] = wlist[e * T_TOK + g]; }
        else           { Tok[tid] = -1;                  Wt[tid] = 0.0f; }
    }
    __syncthreads();

    // ---- loop-invariant staging setup: thread stages 8 contiguous k of one row
    const int sm = tid >> 2;                 // row 0..63
    const int sk = (tid & 3) << 3;           // k offset 0,8,16,24
    const int tokm = Tok[sm];
    const float* arow = x + (size_t)(tokm < 0 ? 0 : tokm) * D_MODEL + sk;
    const float  amask = (tokm >= 0) ? 1.0f : 0.0f;
    const float* brow = w1 + ((size_t)e * HIDDEN + n0 + sm) * D_MODEL + sk;

    const int wave = tid >> 5;
    const int lane = tid & 31;
    const int half = lane >> 4;              // 0: lanes 0-15, 1: lanes 16-31
    const int lr   = lane & 15;
    const int nt   = wave & 3;               // N subtile
    const int ar0  = ((wave >> 2) << 4) + lr;   // A row for acc0
    const int ar1  = ar0 + 32;                  // A row for acc1
    const int brr  = (nt << 4) + lr;            // B row
    const int koff = half << 3;

    v8f acc0 = {}, acc1 = {};

    for (int kc = 0; kc < D_MODEL; kc += 32) {
        // ---- stage A: 2 x float4, masked, split into bf16 hi/lo
        {
            float4 a0 = *(const float4*)(arow + kc);
            float4 a1 = *(const float4*)(arow + kc + 4);
            float va[8] = { a0.x, a0.y, a0.z, a0.w, a1.x, a1.y, a1.z, a1.w };
            #pragma unroll
            for (int j = 0; j < 8; ++j) {
                float v = va[j] * amask;
                __bf16 hi = (__bf16)v;
                Ah[sm][sk + j] = hi;
                Al[sm][sk + j] = (__bf16)(v - (float)hi);
            }
        }
        // ---- stage B: 2 x float4, split into bf16 hi/lo
        {
            float4 b0 = *(const float4*)(brow + kc);
            float4 b1v = *(const float4*)(brow + kc + 4);
            float vb[8] = { b0.x, b0.y, b0.z, b0.w, b1v.x, b1v.y, b1v.z, b1v.w };
            #pragma unroll
            for (int j = 0; j < 8; ++j) {
                float v = vb[j];
                __bf16 hi = (__bf16)v;
                Bh[sm][sk + j] = hi;
                Bl[sm][sk + j] = (__bf16)(v - (float)hi);
            }
        }
        __syncthreads();

        // ---- fragments per ISA 16-bit A/B layouts
        v16bf ah0, al0, ah1, al1, bh, bl;
        #pragma unroll
        for (int j = 0; j < 8; ++j) {
            ah0[j]     = Ah[ar0][koff + j];
            ah0[8 + j] = Ah[ar0][koff + 16 + j];
            al0[j]     = Al[ar0][koff + j];
            al0[8 + j] = Al[ar0][koff + 16 + j];
            ah1[j]     = Ah[ar1][koff + j];
            ah1[8 + j] = Ah[ar1][koff + 16 + j];
            al1[j]     = Al[ar1][koff + j];
            al1[8 + j] = Al[ar1][koff + 16 + j];
        }
        #pragma unroll
        for (int j = 0; j < 16; ++j) {
            bh[j] = Bh[brr][(half << 3) + (j & 7) + ((j >> 3) << 4)];
            bl[j] = Bl[brr][(half << 3) + (j & 7) + ((j >> 3) << 4)];
        }

        // bf16x3: xh*wh + xh*wl + xl*wh  (xl*wl ~2^-18, dropped)
        acc0 = __builtin_amdgcn_wmma_f32_16x16x32_bf16(false, ah0, false, bh, (short)0, acc0, false, false);
        acc0 = __builtin_amdgcn_wmma_f32_16x16x32_bf16(false, ah0, false, bl, (short)0, acc0, false, false);
        acc0 = __builtin_amdgcn_wmma_f32_16x16x32_bf16(false, al0, false, bh, (short)0, acc0, false, false);
        acc1 = __builtin_amdgcn_wmma_f32_16x16x32_bf16(false, ah1, false, bh, (short)0, acc1, false, false);
        acc1 = __builtin_amdgcn_wmma_f32_16x16x32_bf16(false, ah1, false, bl, (short)0, acc1, false, false);
        acc1 = __builtin_amdgcn_wmma_f32_16x16x32_bf16(false, al1, false, bh, (short)0, acc1, false, false);
        __syncthreads();
    }

    // ---- epilogue: C/D layout M = r + 8*half, N = lane%16
    const int n = n0 + (nt << 4) + lr;
    const float bias = b1[e * HIDDEN + n];
    const int mbase = ((wave >> 2) << 4) + (half << 3);
    #pragma unroll
    for (int r = 0; r < 8; ++r) {
        int m = mbase + r;
        int tok = Tok[m];
        if (tok >= 0)
            atomicAdd(&out[(size_t)tok * HIDDEN + n], (acc0[r] + bias) * Wt[m]);
    }
    #pragma unroll
    for (int r = 0; r < 8; ++r) {
        int m = mbase + 32 + r;
        int tok = Tok[m];
        if (tok >= 0)
            atomicAdd(&out[(size_t)tok * HIDDEN + n], (acc1[r] + bias) * Wt[m]);
    }
}

// ---------------------------------------------------------------------------
extern "C" void kernel_launch(void* const* d_in, const int* in_sizes, int n_in,
                              void* d_out, int out_size, void* d_ws, size_t ws_size,
                              hipStream_t stream) {
    const float* x  = (const float*)d_in[0];   // [1,2048,1024]
    const float* rw = (const float*)d_in[1];   // [8,1024]
    const float* w1 = (const float*)d_in[2];   // [8,4096,1024]
    const float* b1 = (const float*)d_in[3];   // [8,4096]
    float* out = (float*)d_out;                // [2048*4096] + l_bal scalar

    // workspace layout
    float* probs = (float*)d_ws;                          // T*E floats
    float* wlist = probs + (size_t)T_TOK * N_EXP;         // E*T floats
    int*   list  = (int*)(wlist + (size_t)N_EXP * T_TOK); // E*T ints
    int*   cnt   = list + (size_t)N_EXP * T_TOK;          // E ints

    moe_zero<<<4096, 256, 0, stream>>>(out, T_TOK * HIDDEN, cnt);
    moe_router<<<T_TOK, 256, 0, stream>>>(x, rw, probs, wlist, list, cnt);
    moe_lbal<<<1, 256, 0, stream>>>(probs, cnt, out + (size_t)T_TOK * HIDDEN);

    dim3 grid(HIDDEN / 64, T_TOK / 64, N_EXP);
    moe_gemm<<<grid, 256, 0, stream>>>(x, w1, b1, list, wlist, cnt, out);
}
